// ISTFT_12017318494878
// MI455X (gfx1250) — compile-verified, matrix operand order checked by reading, not run on hardware
//
#include <hip/hip_runtime.h>
#include <hip/hip_bf16.h>
#include <math.h>

// ---------------------------------------------------------------------------
// iSTFT for MI455X (gfx1250).
//   1) prepass: spectrogram fp32 -> bf16 hi/lo split, stored in WMMA A-fragment
//      layout (one conversion pass instead of per-tile).
//   2) basis init: irfft cos/sin bases (K padded 513->544) as bf16 hi/lo in
//      WMMA B-fragment layout (4.4 MB, L2-resident).
//   3) GEMM: pure v_wmma_f32_16x16x32_bf16 (bf16x3 split = fp32-class
//      accuracy), no LDS / no barriers, 24 WMMAs per K-step per wave.
//   4) deterministic gather-based overlap-add (hop = NFFT/4 -> <=4 frames).
// ---------------------------------------------------------------------------

typedef __attribute__((ext_vector_type(16))) __bf16 v16bf;
typedef __attribute__((ext_vector_type(8)))  float  v8f;

#define NFFT   1024
#define HOP    256
#define NF     513            // rfft bins
#define NSTEP  17             // K padded to 17*32 = 544
#define NT     1024           // frames (T)
#define NBC    64             // batch*channel
#define OUTP   262400         // per-bc output samples: NFFT + HOP*(NT-1) - NFFT/2

static __device__ __forceinline__ unsigned short f2bf(float f) {
    unsigned u = __float_as_uint(f);
    unsigned r = u + 0x7FFFu + ((u >> 16) & 1u);     // round-to-nearest-even
    return (unsigned short)(r >> 16);
}
static __device__ __forceinline__ float bf2f(unsigned short h) {
    return __uint_as_float(((unsigned)h) << 16);
}
// fragment element -> K offset within a 32-K step (16-bit A/B layout):
// lane<16 -> K in {0..7}U{16..23}, lane>=16 -> +8
static __device__ __forceinline__ int frag_k(int lane, int e) {
    return ((lane < 16) ? 0 : 8) + ((e < 8) ? e : e + 8);
}

// ---------------------------------------------------------------------------
// Kernel 1: DFT bases in B-fragment layout.
// Wc[k][n] = wk/32*cos(2pi k n/N), Ws[k][n] = -wk/32*sin(2pi k n/N)
// layout (ushort): [step][n16][which(CH,CL,SH,SL)][lane(32)][e(16)]
// ---------------------------------------------------------------------------
__global__ void istft_basis_init(unsigned short* __restrict__ bfr,
                                 float* __restrict__ win) {
    int gid  = blockIdx.x * blockDim.x + threadIdx.x;   // NSTEP*64*2*32 threads
    int lane = gid & 31;
    int pair = (gid >> 5) & 1;                          // 0 = cos, 1 = sin
    int n16  = (gid >> 6) & 63;
    int step = gid >> 12;
    if (step < NSTEP) {
        int N = n16 * 16 + (lane & 15);
        union { unsigned short u[16]; uint4 q[2]; } H, L;
        #pragma unroll
        for (int e = 0; e < 16; ++e) {
            int K = step * 32 + frag_k(lane, e);
            float v = 0.0f;
            if (K < NF) {
                float wk    = (K == 0 || K == NF - 1) ? 1.0f : 2.0f;
                float scale = wk * (1.0f / 32.0f);      // sqrt(N)/N
                int   m     = (K * N) & (NFFT - 1);     // exact phase reduction
                float ang   = (float)m * (6.283185307179586f / (float)NFFT);
                float sv, cv;
                __sincosf(ang, &sv, &cv);
                v = pair ? (-scale * sv) : (scale * cv);
            }
            unsigned short h = f2bf(v);
            H.u[e] = h;
            L.u[e] = f2bf(v - bf2f(h));
        }
        size_t base = ((((size_t)step * 64 + n16) * 4) + pair * 2) * 512 + lane * 16;
        uint4* dh = (uint4*)(bfr + base);
        dh[0] = H.q[0]; dh[1] = H.q[1];
        uint4* dl = (uint4*)(bfr + base + 512);
        dl[0] = L.q[0]; dl[1] = L.q[1];
    }
    if (gid < NFFT) {                                   // hann * (1/coeff = 0.5)
        float ang = (float)gid * (6.283185307179586f / (float)NFFT);
        win[gid] = 0.25f * (1.0f - __cosf(ang));
    }
}

// ---------------------------------------------------------------------------
// Kernel 2: spectrogram -> bf16 hi/lo A-fragments (single conversion pass).
// layout (ushort): [bc][t16(64)][step][which(ReH,ReL,ImH,ImL)][lane(32)][e(16)]
// ---------------------------------------------------------------------------
__global__ void istft_afrag_prep(const float* __restrict__ x,   // (NBC,NF,NT,2)
                                 unsigned short* __restrict__ afr) {
    int gid  = blockIdx.x * blockDim.x + threadIdx.x;   // NBC*64*NSTEP*2*32
    int lane = gid & 31;
    int pair = (gid >> 5) & 1;                          // 0 = Re, 1 = Im
    int rest = gid >> 6;
    int step = rest % NSTEP;
    rest    /= NSTEP;
    int t16  = rest & 63;
    int bc   = rest >> 6;

    int t = t16 * 16 + (lane & 15);
    union { unsigned short u[16]; uint4 q[2]; } H, L;
    #pragma unroll
    for (int e = 0; e < 16; ++e) {
        int K = step * 32 + frag_k(lane, e);
        float v = 0.0f;
        if (K < NF)                                     // zero-pad K tail
            v = x[(((size_t)bc * NF + K) * NT + t) * 2 + pair];
        unsigned short h = f2bf(v);
        H.u[e] = h;
        L.u[e] = f2bf(v - bf2f(h));
    }
    size_t base = ((((size_t)(bc * 64 + t16) * NSTEP + step) * 4) + pair * 2) * 512 + lane * 16;
    uint4* dh = (uint4*)(afr + base);
    dh[0] = H.q[0]; dh[1] = H.q[1];
    uint4* dl = (uint4*)(afr + base + 512);
    dl[0] = L.q[0]; dl[1] = L.q[1];
}

// ---------------------------------------------------------------------------
// Kernel 3: frames[bc][t][n] = win[n]*(Re . Wc + Im . Ws)   via WMMA.
// Wave owns 16(t) x 64(n): 4 f32 accumulators, 24 WMMAs per 32-K step.
// No LDS, no barriers; fragments come straight from L2-friendly layouts.
// ---------------------------------------------------------------------------
__global__ void __launch_bounds__(256)
istft_irfft_wmma(const unsigned short* __restrict__ afr,
                 const unsigned short* __restrict__ bfr,
                 const float* __restrict__ win,
                 float* __restrict__ frames) {               // (NBC, NT, NFFT)
    const int tid  = threadIdx.x;
    const int lane = tid & 31;
    const int wave = tid >> 5;                   // 0..7
    const int wg   = blockIdx.x;                 // NBC*64*2 blocks
    const int half = wg & 1;
    const int t16  = (wg >> 1) & 63;
    const int bc   = wg >> 7;

    const int nBase = half * 512 + wave * 64;    // wave's 64-wide N span
    const int n16b  = nBase >> 4;

    // uint4 strides: which-block = 64 uint4 (32 lanes x 32B), step-block = 256
    const uint4* abase = (const uint4*)afr +
                         (size_t)(bc * 64 + t16) * (NSTEP * 256) + lane * 2;
    const uint4* bbase = (const uint4*)bfr + lane * 2;

    union Frag { v16bf v; uint4 q[2]; };
    v8f acc[4] = {{}, {}, {}, {}};

    for (int step = 0; step < NSTEP; ++step) {
        const uint4* ap = abase + (size_t)step * 256;
        if (step + 1 < NSTEP)                     // gfx1250 global_prefetch path
            __builtin_prefetch((const void*)(ap + 256), 0, 0);

        Frag aRH, aRL, aIH, aIL;
        aRH.q[0] = ap[0];   aRH.q[1] = ap[1];
        aRL.q[0] = ap[64];  aRL.q[1] = ap[65];
        aIH.q[0] = ap[128]; aIH.q[1] = ap[129];
        aIL.q[0] = ap[192]; aIL.q[1] = ap[193];

        #pragma unroll
        for (int j = 0; j < 4; ++j) {
            const uint4* bp = bbase + ((size_t)step * 64 + (n16b + j)) * 256;
            Frag bCH, bCL, bSH, bSL;
            bCH.q[0] = bp[0];   bCH.q[1] = bp[1];
            bCL.q[0] = bp[64];  bCL.q[1] = bp[65];
            bSH.q[0] = bp[128]; bSH.q[1] = bp[129];
            bSL.q[0] = bp[192]; bSL.q[1] = bp[193];
            // bf16x3: ah*bh + ah*bl + al*bh, f32 accumulate (Re and Im terms)
            acc[j] = __builtin_amdgcn_wmma_f32_16x16x32_bf16(false, aRH.v, false, bCH.v, (short)0, acc[j], false, false);
            acc[j] = __builtin_amdgcn_wmma_f32_16x16x32_bf16(false, aRH.v, false, bCL.v, (short)0, acc[j], false, false);
            acc[j] = __builtin_amdgcn_wmma_f32_16x16x32_bf16(false, aRL.v, false, bCH.v, (short)0, acc[j], false, false);
            acc[j] = __builtin_amdgcn_wmma_f32_16x16x32_bf16(false, aIH.v, false, bSH.v, (short)0, acc[j], false, false);
            acc[j] = __builtin_amdgcn_wmma_f32_16x16x32_bf16(false, aIH.v, false, bSL.v, (short)0, acc[j], false, false);
            acc[j] = __builtin_amdgcn_wmma_f32_16x16x32_bf16(false, aIL.v, false, bSH.v, (short)0, acc[j], false, false);
        }
    }

    // epilogue: C/D layout: VGPR r -> M = r + 8*(lane>=16); N = lane&15
    const int rowSel = (lane >> 4) << 3;
    const int nl     = lane & 15;
    #pragma unroll
    for (int j = 0; j < 4; ++j) {
        const int   n  = nBase + j * 16 + nl;
        const float wv = win[n];
        float* dst = frames + ((size_t)bc * NT + (t16 * 16 + rowSel)) * NFFT + n;
        #pragma unroll
        for (int r = 0; r < 8; ++r)
            dst[(size_t)r * NFFT] = acc[j][r] * wv;
    }
}

// ---------------------------------------------------------------------------
// Kernel 4: deterministic overlap-add (gather <=4 contributing frames),
// with the NFFT/2 crop folded in.
// ---------------------------------------------------------------------------
__global__ void istft_overlap_add(const float* __restrict__ frames,
                                  float* __restrict__ out) {
    int o = blockIdx.x * blockDim.x + threadIdx.x;
    if (o >= NBC * OUTP) return;
    int bc = o / OUTP;
    int s  = o - bc * OUTP;
    int p  = s + NFFT / 2;
    int jmin = (p >= NFFT - HOP) ? ((p - (NFFT - HOP)) >> 8) : 0;
    int jmax = p >> 8; if (jmax > NT - 1) jmax = NT - 1;
    const float* fb = frames + (size_t)bc * NT * NFFT;
    float sum = 0.0f;
    for (int j = jmin; j <= jmax; ++j)
        sum += fb[(size_t)j * NFFT + (p - (j << 8))];
    out[o] = sum;
}

// ---------------------------------------------------------------------------
extern "C" void kernel_launch(void* const* d_in, const int* in_sizes, int n_in,
                              void* d_out, int out_size, void* d_ws, size_t ws_size,
                              hipStream_t stream) {
    (void)in_sizes; (void)n_in; (void)out_size; (void)ws_size;
    const float* x   = (const float*)d_in[0];
    float*       out = (float*)d_out;
    char*        ws  = (char*)d_ws;

    // workspace layout:
    //   frames fp32:  NBC*NT*NFFT                  = 256 MiB
    //   A-frags bf16: NBC*64*NSTEP*4*512 ushorts   = 272 MiB
    //   B-frags bf16: NSTEP*64*4*512 ushorts       = 4.25 MiB
    //   win fp32:     NFFT
    const size_t FR = (size_t)NBC * NT * NFFT * sizeof(float);
    const size_t AF = (size_t)NBC * 64 * NSTEP * 4 * 512 * sizeof(unsigned short);
    const size_t BF = (size_t)NSTEP * 64 * 4 * 512 * sizeof(unsigned short);
    float*          frames = (float*)ws;
    unsigned short* afr    = (unsigned short*)(ws + FR);
    unsigned short* bfr    = (unsigned short*)(ws + FR + AF);
    float*          win    = (float*)(ws + FR + AF + BF);

    istft_basis_init<<<(NSTEP * 64 * 2 * 32) / 256, 256, 0, stream>>>(bfr, win);
    istft_afrag_prep<<<(NBC * 64 * NSTEP * 2 * 32) / 256, 256, 0, stream>>>(x, afr);
    istft_irfft_wmma<<<NBC * 64 * 2, 256, 0, stream>>>(afr, bfr, win, frames);
    istft_overlap_add<<<(NBC * OUTP + 255) / 256, 256, 0, stream>>>(frames, out);
}
